// DLSPooling_1632087572809
// MI455X (gfx1250) — compile-verified, matrix operand order checked by reading, not run on hardware
//
#include <hip/hip_runtime.h>
#include <hip/hip_bf16.h>

// ---------------------------------------------------------------------------
// DLSPooling for MI455X (gfx1250, wave32, WMMA).
// Stages:
//   k_node_gemm : m = elu(x@Wm), acc = x@Wr + b    (bf16 WMMA 16x16x32, W in LDS)
//   k_xt        : xt[b][c][n] = bf16(x)            (transpose staging for pool GEMM)
//   k_edges     : acc[dst] += m[src]               (f32 atomics, L2-resident)
//   k_softmax   : St[b][k][n] = softmax(tanh(acc)) (wave32 shuffle reduction, bf16)
//   k_pool      : out[b] = St_b @ xt_b^T           (bf16 WMMA 16x16x32)
//   k_tail      : edge_index_out / batch_out       (pure index arithmetic)
// ---------------------------------------------------------------------------

typedef __attribute__((ext_vector_type(16))) __bf16 v16bf;
typedef __attribute__((ext_vector_type(8)))  __bf16 v8bf;
typedef __attribute__((ext_vector_type(8)))  float  v8f;

#define NB   64
#define NPER 1024
#define CIN  128
#define KCL  64
#define NTOT (NB * NPER)

// Build a 16-bit A/B WMMA fragment from a contiguous bf16 row segment.
// Per ISA 7.12.2: lane half h=lane>>4 selects k-offset h*8; elements 0..7 are
// k = h*8 + 0..7 and elements 8..15 are k = 16 + h*8 + 0..7 (p already
// includes the +h*8).
static __device__ __forceinline__ v16bf load_frag_bf16(const __bf16* p) {
  v16bf a;
  *reinterpret_cast<v8bf*>(&a)       = *reinterpret_cast<const v8bf*>(p);
  *(reinterpret_cast<v8bf*>(&a) + 1) = *reinterpret_cast<const v8bf*>(p + 16);
  return a;
}

// Same fragment layout, but sourced from fp32 with on-the-fly bf16 convert.
static __device__ __forceinline__ v16bf make_frag_f32(const float* px) {
  float4 f0 = *(const float4*)(px);
  float4 f1 = *(const float4*)(px + 4);
  float4 f2 = *(const float4*)(px + 16);
  float4 f3 = *(const float4*)(px + 20);
  v16bf a;
  a[0]  = (__bf16)f0.x; a[1]  = (__bf16)f0.y; a[2]  = (__bf16)f0.z; a[3]  = (__bf16)f0.w;
  a[4]  = (__bf16)f1.x; a[5]  = (__bf16)f1.y; a[6]  = (__bf16)f1.z; a[7]  = (__bf16)f1.w;
  a[8]  = (__bf16)f2.x; a[9]  = (__bf16)f2.y; a[10] = (__bf16)f2.z; a[11] = (__bf16)f2.w;
  a[12] = (__bf16)f3.x; a[13] = (__bf16)f3.y; a[14] = (__bf16)f3.z; a[15] = (__bf16)f3.w;
  return a;
}

// ---- Stage 1: fused node GEMMs: m = elu(x@Wm), acc = x@Wr + b --------------
// One wave per 16-row tile of x. Both weight matrices staged in LDS in
// B-fragment order: [mat][ks*4+ct][lane][16 half].
__global__ void __launch_bounds__(256)
k_node_gemm(const float* __restrict__ x, const float* __restrict__ Wm,
            const float* __restrict__ Wr, const float* __restrict__ bias,
            float* __restrict__ m_buf, float* __restrict__ acc_buf)
{
  __shared__ __align__(32) __bf16 ldsW[2 * 16 * 32 * 16];  // 32 KB
  const int tid = threadIdx.x;

  // Swizzle W (128x64 row-major fp32) into WMMA B-fragment layout, bf16.
  for (int idx = tid; idx < 8192; idx += 256) {
    int f = idx >> 9, l = (idx >> 4) & 31, i = idx & 15;
    int n = l & 15, h = l >> 4, j = i >> 1, s = i & 1;
    int kk = (j < 4) ? (h * 8 + 2 * j + s) : (16 + h * 8 + 2 * (j - 4) + s);
    int ks = f >> 2, ct = f & 3;
    int widx = (ks * 32 + kk) * KCL + ct * 16 + n;
    ldsW[idx]        = (__bf16)Wm[widx];
    ldsW[8192 + idx] = (__bf16)Wr[widx];
  }
  __syncthreads();

  const int wid = tid >> 5, lane = tid & 31;
  const int tile = blockIdx.x * 8 + wid;          // 4096 tiles of 16 rows
  const int r0 = tile * 16;
  const int lrow = lane & 15, h = lane >> 4;

  v8f accM[4] = {};
  v8f accR[4] = {};

  const float* xrow = x + (size_t)(r0 + lrow) * CIN;
#pragma unroll
  for (int ks = 0; ks < 4; ++ks) {                // reduce over C=128 in 4x32
    v16bf a = make_frag_f32(xrow + ks * 32 + h * 8);
#pragma unroll
    for (int ct = 0; ct < 4; ++ct) {              // 4 column tiles of 16
      v16bf bm = *reinterpret_cast<const v16bf*>(&ldsW[((ks * 4 + ct) * 32 + lane) * 16]);
      accM[ct] = __builtin_amdgcn_wmma_f32_16x16x32_bf16(
          false, a, false, bm, (short)0, accM[ct], false, false);
      v16bf br = *reinterpret_cast<const v16bf*>(&ldsW[8192 + ((ks * 4 + ct) * 32 + lane) * 16]);
      accR[ct] = __builtin_amdgcn_wmma_f32_16x16x32_bf16(
          false, a, false, br, (short)0, accR[ct], false, false);
    }
  }

  // D layout: vgpr v, lane -> (row = v + 8*(lane>>4), col = lane&15)
#pragma unroll
  for (int ct = 0; ct < 4; ++ct) {
#pragma unroll
    for (int v = 0; v < 8; ++v) {
      int grow = r0 + h * 8 + v;
      int gcol = ct * 16 + lrow;
      float mv = accM[ct][v];
      m_buf[(size_t)grow * KCL + gcol]   = (mv > 0.f) ? mv : (__expf(mv) - 1.f);  // ELU
      acc_buf[(size_t)grow * KCL + gcol] = accR[ct][v] + bias[gcol];
    }
  }
}

// ---- Stage 1b: xt[b][c][n] = bf16(x[n][c]) ---------------------------------
__global__ void __launch_bounds__(256)
k_xt(const float* __restrict__ x, __bf16* __restrict__ xt)
{
  int gid = blockIdx.x * 256 + threadIdx.x;       // NTOT*32 threads
  int n = gid >> 5, c4 = (gid & 31) * 4;
  float4 v = *(const float4*)(x + (size_t)n * CIN + c4);
  int b = n >> 10, nl = n & 1023;
  __bf16* base = xt + ((size_t)b * CIN + c4) * NPER + nl;
  base[0]        = (__bf16)v.x;
  base[NPER]     = (__bf16)v.y;
  base[2 * NPER] = (__bf16)v.z;
  base[3 * NPER] = (__bf16)v.w;
}

// ---- Stage 2: acc[dst] += m[src] over all edges (f32 atomics, L2-resident) -
__global__ void __launch_bounds__(256)
k_edges(const int* __restrict__ ei, const float* __restrict__ m_buf,
        float* __restrict__ acc_buf, int E)
{
  int gid = blockIdx.x * 256 + threadIdx.x;       // E*16 threads
  int e = gid >> 4, kq = (gid & 15) * 4;
  int src = ei[e];
  int dst = ei[E + e];
  float4 mv = *(const float4*)(m_buf + (size_t)src * KCL + kq);
  float* d = acc_buf + (size_t)dst * KCL + kq;
  atomicAdd(d + 0, mv.x);
  atomicAdd(d + 1, mv.y);
  atomicAdd(d + 2, mv.z);
  atomicAdd(d + 3, mv.w);
}

// ---- Stage 3: St[b][k][n] = softmax_k(tanh(acc[n][k])) as bf16 -------------
// One wave32 per node; each lane owns clusters 2*lane, 2*lane+1.
__global__ void __launch_bounds__(256)
k_softmax(const float* __restrict__ acc_buf, __bf16* __restrict__ St)
{
  int wid = threadIdx.x >> 5, lane = threadIdx.x & 31;
  int n = blockIdx.x * 8 + wid;
  float2 v = *(const float2*)(acc_buf + (size_t)n * KCL + lane * 2);
  float a0 = tanhf(v.x), a1 = tanhf(v.y);
  float mx = fmaxf(a0, a1);
#pragma unroll
  for (int off = 16; off > 0; off >>= 1) mx = fmaxf(mx, __shfl_xor(mx, off, 32));
  float e0 = __expf(a0 - mx), e1 = __expf(a1 - mx);
  float sm = e0 + e1;
#pragma unroll
  for (int off = 16; off > 0; off >>= 1) sm += __shfl_xor(sm, off, 32);
  float inv = 1.0f / sm;
  int b = n >> 10, nl = n & 1023;
  __bf16* dst = St + ((size_t)b * KCL + lane * 2) * NPER + nl;  // transposed store
  dst[0]    = (__bf16)(e0 * inv);
  dst[NPER] = (__bf16)(e1 * inv);
}

// ---- Stage 4: out[b] = S_b^T X_b  ([64x1024]@[1024x128]) via bf16 WMMA -----
// One block per graph, 8 waves: wave owns (mt = wid>>1, ct0 = (wid&1)*4).
__global__ void __launch_bounds__(256)
k_pool(const __bf16* __restrict__ St, const __bf16* __restrict__ xt,
       float* __restrict__ xout)
{
  const int b = blockIdx.x;
  const int tid = threadIdx.x, wid = tid >> 5, lane = tid & 31;
  const int mt = wid >> 1, ct0 = (wid & 1) * 4;
  const int lrow = lane & 15, h = lane >> 4;

  const __bf16* sbase = St + ((size_t)b * KCL + mt * 16 + lrow) * NPER;
  v8f acc[4] = {};
#pragma unroll 4
  for (int ks = 0; ks < 32; ++ks) {               // reduce over 1024 nodes
    int nb = ks * 32;
    v16bf a = load_frag_bf16(sbase + nb + h * 8);
#pragma unroll
    for (int ct = 0; ct < 4; ++ct) {
      const __bf16* xb = xt + ((size_t)b * CIN + (ct0 + ct) * 16 + lrow) * NPER + nb + h * 8;
      v16bf bf = load_frag_bf16(xb);
      acc[ct] = __builtin_amdgcn_wmma_f32_16x16x32_bf16(
          false, a, false, bf, (short)0, acc[ct], false, false);
    }
  }
#pragma unroll
  for (int ct = 0; ct < 4; ++ct)
#pragma unroll
    for (int v = 0; v < 8; ++v)
      xout[((size_t)b * KCL + mt * 16 + h * 8 + v) * CIN + (ct0 + ct) * 16 + lrow] = acc[ct][v];
}

// ---- Stage 5: edge_index_out / batch_out (index arithmetic only) -----------
// Integer-valued outputs written as exact small floats into the float d_out.
__global__ void __launch_bounds__(256)
k_tail(float* __restrict__ out)
{
  int idx = blockIdx.x * 256 + threadIdx.x;
  const int EOUT = NB * KCL * KCL;                // 262144 per row
  if (idx < 2 * EOUT) {
    int r = idx / EOUT;
    int j = idx - r * EOUT;
    int b = j / (KCL * KCL), p = j % (KCL * KCL);
    int val = ((r == 0) ? (p / KCL) : (p % KCL)) + b * KCL;
    out[NB * KCL * CIN + idx] = (float)val;
  } else {
    int i = idx - 2 * EOUT;
    if (i < NB * KCL)
      out[NB * KCL * CIN + 2 * EOUT + i] = (float)(i / KCL);
  }
}

extern "C" void kernel_launch(void* const* d_in, const int* in_sizes, int n_in,
                              void* d_out, int out_size, void* d_ws, size_t ws_size,
                              hipStream_t stream)
{
  const float* x    = (const float*)d_in[0];
  const int*   ei   = (const int*)d_in[1];
  // d_in[2] = batch (unused; graphs are equal-sized)
  const float* Wm   = (const float*)d_in[3];
  const float* Wr   = (const float*)d_in[4];
  const float* bias = (const float*)d_in[5];
  const int E = in_sizes[1] / 2;                  // 2,097,152

  // Workspace layout: m (16MB) | acc (16MB) | St bf16 (8MB) | xt bf16 (16MB)
  char* ws = (char*)d_ws;
  float*  m_buf = (float*)ws;
  float*  acc   = (float*)(ws + (size_t)NTOT * KCL * 4);
  __bf16* St    = (__bf16*)(ws + (size_t)2 * NTOT * KCL * 4);
  __bf16* xt    = (__bf16*)(ws + (size_t)2 * NTOT * KCL * 4 + (size_t)NTOT * KCL * 2);

  k_node_gemm<<<NTOT / 16 / 8, 256, 0, stream>>>(x, Wm, Wr, bias, m_buf, acc);
  k_xt<<<(NTOT * 32) / 256, 256, 0, stream>>>(x, xt);
  k_edges<<<(unsigned)((size_t)E * 16 / 256), 256, 0, stream>>>(ei, m_buf, acc, E);
  k_softmax<<<NTOT / 8, 256, 0, stream>>>(acc, St);
  k_pool<<<NB, 256, 0, stream>>>(St, xt, (float*)d_out);
  k_tail<<<(2 * NB * KCL * KCL + NB * KCL) / 256, 256, 0, stream>>>((float*)d_out);
}
